// GAT_class_64295660421705
// MI455X (gfx1250) — matrix operand order repaired and checked by reading, not compiled, not run on hardware
//
#include <hip/hip_runtime.h>
#include <hip/hip_bf16.h>

// ---------------- problem constants (from reference) ----------------
enum : int { NN = 20000, FF = 128, HH = 8, CC = 64, HCC = 512, EDD = 4,
             EE = 320000, GG = 200, EAA = EE + NN };

typedef __attribute__((ext_vector_type(16))) __bf16 v16bf;
typedef __attribute__((ext_vector_type(8)))  float  v8f;

// ---------------- small helpers ----------------
__device__ __forceinline__ unsigned f2key(float f) {
  unsigned b = __float_as_uint(f);
  return (b & 0x80000000u) ? ~b : (b | 0x80000000u);
}
__device__ __forceinline__ float key2f(unsigned k) {
  return (k & 0x80000000u) ? __uint_as_float(k & 0x7fffffffu)
                           : __uint_as_float(~k);
}

__global__ void k_fill(float* __restrict__ p, float v, long n) {
  long i = (long)blockIdx.x * 256 + threadIdx.x;
  if (i < n) p[i] = v;
}

__global__ void k_f2bf(const float* __restrict__ in, __bf16* __restrict__ out, long n) {
  long i = (long)blockIdx.x * 256 + threadIdx.x;
  if (i < n) out[i] = (__bf16)in[i];
}

// transposed bf16 weights: out[n*K + k] = (bf16) in[k*Nn + n]
__global__ void k_f2bf_t(const float* __restrict__ in, __bf16* __restrict__ out,
                         int K, int Nn) {
  long i = (long)blockIdx.x * 256 + threadIdx.x;
  if (i >= (long)K * Nn) return;
  int n = (int)(i / K), k = (int)(i % K);
  out[i] = (__bf16)in[(long)k * Nn + n];
}

// ---------------- self-loop edge_attr = mean of incoming ----------------
__global__ void k_deg_emean(const int* __restrict__ ei, const float* __restrict__ eattr,
                            float* __restrict__ deg, float* __restrict__ emean) {
  int e = blockIdx.x * 256 + threadIdx.x;
  if (e >= EE) return;
  int d = ei[EE + e];
  atomicAdd(&deg[d], 1.0f);
#pragma unroll
  for (int k = 0; k < EDD; ++k)
    atomicAdd(&emean[(long)d * EDD + k], eattr[(long)e * EDD + k]);
}

__global__ void k_emean_div(float* __restrict__ emean, const float* __restrict__ deg) {
  int i = blockIdx.x * 256 + threadIdx.x;
  if (i >= NN * EDD) return;
  emean[i] /= fmaxf(deg[i >> 2], 1.0f);
}

// ---------------- WMMA bf16 GEMM with async global->LDS staging ----------------
// C[M,Nn] = A[M,K] @ W^T  where Bt_w is the transposed weight Wt[Nn][K] (bf16).
#define TMB 128  // block tile M (4 waves x 32 rows)
#define TNB 64   // block tile N
#define TKB 32   // K chunk (one wmma K per chunk)

union BF16x16 { v16bf v; uint4 q[2]; __bf16 h[16]; };

__device__ __forceinline__ void async_b128(unsigned lds_off, const __bf16* g) {
  // GLOBAL_LOAD_ASYNC_TO_LDS_B128 (ASYNCcnt) — per-lane 16B Memory->LDS copy
  asm volatile("global_load_async_to_lds_b128 %0, %1, off"
               :: "v"(lds_off), "v"((unsigned long long)(uintptr_t)g)
               : "memory");
}
__device__ __forceinline__ void wait_async0() {
#if __has_builtin(__builtin_amdgcn_s_wait_asynccnt)
  __builtin_amdgcn_s_wait_asynccnt(0);
#else
  asm volatile("s_wait_asynccnt 0x0" ::: "memory");
#endif
}
__device__ __forceinline__ unsigned lds_off32(const void* p) {
  return (unsigned)(uintptr_t)p;  // LDS addr = generic addr[31:0] (ISA 10.2)
}

__global__ __launch_bounds__(128)
void k_gemm_bf16(const __bf16* __restrict__ A, const __bf16* __restrict__ Wt,
                 float* __restrict__ C, int M, int Nn, int K) {
  __shared__ __bf16 At[TMB][TKB];  // A tile, row-major
  __shared__ __bf16 Bt[TNB][TKB];  // Wt tile: Bt[n][k]
  const int tid  = threadIdx.x;
  const int wave = tid >> 5;
  const int lane = tid & 31;
  const int tile_n = blockIdx.x * TNB;
  const int tile_m = blockIdx.y * TMB;

  v8f acc[2][4];
#pragma unroll
  for (int mi = 0; mi < 2; ++mi)
#pragma unroll
    for (int ns = 0; ns < 4; ++ns) acc[mi][ns] = (v8f)(0.0f);

  // Per-thread staging chunks: A = 4x16B, B = 2x16B (no branches: rows clamped)
  unsigned ldsA[4]; const __bf16* gA[4];
#pragma unroll
  for (int i = 0; i < 4; ++i) {
    int c = tid + i * 128;            // 512 chunks: row = c/4, col = (c%4)*8
    int row = c >> 2, coff = (c & 3) * 8;
    int gm = tile_m + row; gm = gm < (M - 1) ? gm : (M - 1);
    ldsA[i] = lds_off32(&At[row][coff]);
    gA[i]   = A + (long)gm * K + coff;
  }
  unsigned ldsB[2]; const __bf16* gB[2];
#pragma unroll
  for (int i = 0; i < 2; ++i) {
    int c = tid + i * 128;            // 256 chunks
    int n = c >> 2, coff = (c & 3) * 8;
    ldsB[i] = lds_off32(&Bt[n][coff]);
    gB[i]   = Wt + (long)(tile_n + n) * K + coff;
  }

  const int mrow = wave * 32 + (lane & 15);
  const int ks   = (lane >> 4) * 8;
  const int kb   = (lane >> 4) * 16;

  for (int k0 = 0; k0 < K; k0 += TKB) {
#pragma unroll
    for (int i = 0; i < 4; ++i) { async_b128(ldsA[i], gA[i]); gA[i] += TKB; }
#pragma unroll
    for (int i = 0; i < 2; ++i) { async_b128(ldsB[i], gB[i]); gB[i] += TKB; }
    wait_async0();
    __syncthreads();

    BF16x16 a0, a1;
    a0.q[0] = *(const uint4*)&At[mrow][ks];
    a0.q[1] = *(const uint4*)&At[mrow][16 + ks];
    a1.q[0] = *(const uint4*)&At[mrow + 16][ks];
    a1.q[1] = *(const uint4*)&At[mrow + 16][16 + ks];
#pragma unroll
    for (int ns = 0; ns < 4; ++ns) {
      BF16x16 b;
      const int bn = ns * 16 + (lane & 15);
      b.q[0] = *(const uint4*)&Bt[bn][ks];
      b.q[1] = *(const uint4*)&Bt[bn][16 + ks];
      acc[0][ns] = __builtin_amdgcn_wmma_f32_16x16x32_bf16(
          false, a0.v, false, b.v, (short)0, acc[0][ns], false, false);
      acc[1][ns] = __builtin_amdgcn_wmma_f32_16x16x32_bf16(
          false, a1.v, false, b.v, (short)0, acc[1][ns], false, false);
    }
    __syncthreads();
  }

  const int col  = lane & 15;
  const int rsel = (lane >> 4) * 8;
#pragma unroll
  for (int mi = 0; mi < 2; ++mi)
#pragma unroll
    for (int ns = 0; ns < 4; ++ns)
#pragma unroll
      for (int j = 0; j < 8; ++j) {
        int gm = tile_m + wave * 32 + mi * 16 + rsel + j;
        if (gm < M) C[(long)gm * Nn + tile_n + ns * 16 + col] = acc[mi][ns][j];
      }
}

// ---------------- attention precompute ----------------
__global__ void k_attM(const float* __restrict__ We, const float* __restrict__ att_e,
                       float* __restrict__ attM) {
  int t = threadIdx.x;
  if (t >= EDD * HH) return;
  int k = t >> 3, h = t & 7;
  float s = 0.0f;
  for (int c = 0; c < CC; ++c) s += We[(long)k * HCC + h * CC + c] * att_e[h * CC + c];
  attM[k * HH + h] = s;
}

__global__ void k_as_ad(const float* __restrict__ hbuf, const float* __restrict__ att_s,
                        const float* __restrict__ att_d, float* __restrict__ asb,
                        float* __restrict__ adb) {
  int idx = blockIdx.x * 256 + threadIdx.x;
  if (idx >= NN * HH) return;
  int n = idx >> 3, h = idx & 7;
  const float* hp = hbuf + (long)n * HCC + h * CC;
  const float* sp = att_s + h * CC;
  const float* dp = att_d + h * CC;
  float s1 = 0.f, s2 = 0.f;
#pragma unroll 4
  for (int c = 0; c < CC; c += 4) {
    float4 hv = *(const float4*)(hp + c);
    float4 sv = *(const float4*)(sp + c);
    float4 dv = *(const float4*)(dp + c);
    s1 += hv.x * sv.x + hv.y * sv.y + hv.z * sv.z + hv.w * sv.w;
    s2 += hv.x * dv.x + hv.y * dv.y + hv.z * dv.z + hv.w * dv.w;
  }
  asb[idx] = s1;
  adb[idx] = s2;
}

__global__ void k_alpha(const int* __restrict__ ei, const float* __restrict__ eattr,
                        const float* __restrict__ emean, const float* __restrict__ asb,
                        const float* __restrict__ adb, const float* __restrict__ attM,
                        float* __restrict__ alpha, unsigned* __restrict__ amaxK) {
  long e = (long)blockIdx.x * 256 + threadIdx.x;
  if (e >= EAA) return;
  int s, d;
  float ev[EDD];
  if (e < EE) {
    s = ei[e]; d = ei[EE + e];
#pragma unroll
    for (int k = 0; k < EDD; ++k) ev[k] = eattr[e * EDD + k];
  } else {
    s = d = (int)(e - EE);
#pragma unroll
    for (int k = 0; k < EDD; ++k) ev[k] = emean[(long)s * EDD + k];
  }
#pragma unroll
  for (int h = 0; h < HH; ++h) {
    float ae = 0.f;
#pragma unroll
    for (int k = 0; k < EDD; ++k) ae += ev[k] * attM[k * HH + h];
    float al = asb[(long)s * HH + h] + adb[(long)d * HH + h] + ae;
    al = (al > 0.f) ? al : 0.2f * al;
    alpha[e * HH + h] = al;
    atomicMax(&amaxK[(long)d * HH + h], f2key(al));
  }
}

__global__ void k_expsum(const int* __restrict__ ei, float* __restrict__ alpha,
                         const unsigned* __restrict__ amaxK, float* __restrict__ denom) {
  long e = (long)blockIdx.x * 256 + threadIdx.x;
  if (e >= EAA) return;
  int d = (e < EE) ? ei[EE + e] : (int)(e - EE);
#pragma unroll
  for (int h = 0; h < HH; ++h) {
    float am = key2f(amaxK[(long)d * HH + h]);
    float v = __expf(alpha[e * HH + h] - am);
    alpha[e * HH + h] = v;
    atomicAdd(&denom[(long)d * HH + h], v);
  }
}

// one wave per edge; lane owns 16 contiguous channels -> head = lane>>2 (lane-constant)
__global__ __launch_bounds__(256)
void k_scatter(const int* __restrict__ ei, const float* __restrict__ hbuf,
               const float* __restrict__ alpha, const float* __restrict__ denom,
               float* __restrict__ outb) {
  const int wave = threadIdx.x >> 5;
  const int lane = threadIdx.x & 31;
  long e = (long)blockIdx.x * 8 + wave;
  if (e >= EAA) return;
  int s, d;
  if (e < EE) { s = ei[e]; d = ei[EE + e]; }
  else        { s = d = (int)(e - EE); }
  const int h = lane >> 2;
  const float w = alpha[e * HH + h] / (denom[(long)d * HH + h] + 1e-16f);
  const float* hp = hbuf + (long)s * HCC + lane * 16;
  float* op = outb + (long)d * HCC + lane * 16;
  __builtin_prefetch(hp, 0, 0);
#pragma unroll
  for (int i = 0; i < 4; ++i) {
    float4 v = *(const float4*)(hp + i * 4);
    atomicAdd(op + i * 4 + 0, v.x * w);
    atomicAdd(op + i * 4 + 1, v.y * w);
    atomicAdd(op + i * 4 + 2, v.z * w);
    atomicAdd(op + i * 4 + 3, v.w * w);
  }
}

__global__ void k_relu_bias_bf(const float* __restrict__ outb, const float* __restrict__ b,
                               __bf16* __restrict__ dst, long n) {
  long i = (long)blockIdx.x * 256 + threadIdx.x;
  if (i >= n) return;
  int c = (int)(i % HCC);
  float v = outb[i] + b[c];
  dst[i] = (__bf16)(v > 0.f ? v : 0.f);
}

__global__ void k_cnt(const int* __restrict__ batch, float* __restrict__ cnt) {
  int n = blockIdx.x * 256 + threadIdx.x;
  if (n < NN) atomicAdd(&cnt[batch[n]], 1.0f);
}

__global__ void k_pool(const float* __restrict__ outb, const float* __restrict__ b2,
                       const int* __restrict__ batch, float* __restrict__ pooled) {
  long i = (long)blockIdx.x * 256 + threadIdx.x;
  if (i >= (long)NN * HCC) return;
  int n = (int)(i / HCC), c = (int)(i % HCC);
  atomicAdd(&pooled[(long)batch[n] * HCC + c], outb[i] + b2[c]);
}

__global__ void k_head(const float* __restrict__ pooled, const float* __restrict__ cnt,
                       const float* __restrict__ lw, const float* __restrict__ lb,
                       float* __restrict__ outp) {
  int i = blockIdx.x * 256 + threadIdx.x;
  if (i >= GG * 2) return;
  int g = i >> 1, o = i & 1;
  float inv = 1.0f / fmaxf(cnt[g], 1.0f);
  float s = 0.0f;
  for (int c = 0; c < HCC; ++c) s += pooled[(long)g * HCC + c] * lw[c * 2 + o];
  outp[i] = s * inv + lb[o];
}

// ---------------- host launcher ----------------
#define LAUNCH(k, n, ...) k<<<dim3((unsigned)(((n) + 255) / 256)), dim3(256), 0, stream>>>(__VA_ARGS__)

extern "C" void kernel_launch(void* const* d_in, const int* in_sizes, int n_in,
                              void* d_out, int out_size, void* d_ws, size_t ws_size,
                              hipStream_t stream) {
  const float* x      = (const float*)d_in[0];
  const int*   ei     = (const int*)d_in[1];
  const int*   batch  = (const int*)d_in[2];
  const float* eattr  = (const float*)d_in[3];
  const float* W1     = (const float*)d_in[4];
  const float* att_s1 = (const float*)d_in[5];
  const float* att_d1 = (const float*)d_in[6];
  const float* att_e1 = (const float*)d_in[7];
  const float* We1    = (const float*)d_in[8];
  const float* b1     = (const float*)d_in[9];
  const float* W2     = (const float*)d_in[10];
  const float* att_s2 = (const float*)d_in[11];
  const float* att_d2 = (const float*)d_in[12];
  const float* att_e2 = (const float*)d_in[13];
  const float* We2    = (const float*)d_in[14];
  const float* b2     = (const float*)d_in[15];
  const float* lin_w  = (const float*)d_in[16];
  const float* lin_b  = (const float*)d_in[17];
  float* outp = (float*)d_out;

  char* w = (char*)d_ws;
  size_t off = 0;
  auto carve = [&](size_t bytes) -> char* {
    char* p = w + off;
    off = (off + bytes + 255) & ~(size_t)255;
    return p;
  };

  float*    deg    = (float*)carve((size_t)NN * 4);
  float*    emean  = (float*)carve((size_t)NN * EDD * 4);
  float*    attM   = (float*)carve((size_t)EDD * HH * 4);
  float*    asb    = (float*)carve((size_t)NN * HH * 4);
  float*    adb    = (float*)carve((size_t)NN * HH * 4);
  unsigned* amaxK  = (unsigned*)carve((size_t)NN * HH * 4);
  float*    denom  = (float*)carve((size_t)NN * HH * 4);
  float*    alpha  = (float*)carve((size_t)EAA * HH * 4);
  float*    hbuf   = (float*)carve((size_t)NN * HCC * 4);
  float*    obuf   = (float*)carve((size_t)NN * HCC * 4);
  float*    pooled = (float*)carve((size_t)GG * HCC * 4);
  float*    cnt    = (float*)carve((size_t)GG * 4);
  __bf16*   Abf    = (__bf16*)carve((size_t)NN * HCC * 2);
  __bf16*   Wbf    = (__bf16*)carve((size_t)HCC * HCC * 2);  // transposed weights

  const dim3 gGemm(HCC / TNB, (NN + TMB - 1) / TMB);

  // ---- self-loop edge_attr (mean of incoming) ----
  LAUNCH(k_fill, NN, deg, 0.f, (long)NN);
  LAUNCH(k_fill, NN * EDD, emean, 0.f, (long)NN * EDD);
  LAUNCH(k_deg_emean, EE, ei, eattr, deg, emean);
  LAUNCH(k_emean_div, NN * EDD, emean, deg);

  // ---- layer 1 ----
  LAUNCH(k_f2bf, NN * FF, x, Abf, (long)NN * FF);
  LAUNCH(k_f2bf_t, FF * HCC, W1, Wbf, FF, HCC);
  k_gemm_bf16<<<gGemm, 128, 0, stream>>>(Abf, Wbf, hbuf, NN, HCC, FF);
  k_attM<<<1, 32, 0, stream>>>(We1, att_e1, attM);
  LAUNCH(k_as_ad, NN * HH, hbuf, att_s1, att_d1, asb, adb);
  LAUNCH(k_fill, NN * HH, (float*)amaxK, 0.f, (long)NN * HH);
  LAUNCH(k_fill, NN * HH, denom, 0.f, (long)NN * HH);
  LAUNCH(k_fill, NN * HCC, obuf, 0.f, (long)NN * HCC);
  LAUNCH(k_alpha, EAA, ei, eattr, emean, asb, adb, attM, alpha, amaxK);
  LAUNCH(k_expsum, EAA, ei, alpha, amaxK, denom);
  k_scatter<<<(EAA + 7) / 8, 256, 0, stream>>>(ei, hbuf, alpha, denom, obuf);
  LAUNCH(k_relu_bias_bf, (long)NN * HCC, obuf, b1, Abf, (long)NN * HCC);

  // ---- layer 2 ----
  LAUNCH(k_f2bf_t, HCC * HCC, W2, Wbf, HCC, HCC);
  k_gemm_bf16<<<gGemm, 128, 0, stream>>>(Abf, Wbf, hbuf, NN, HCC, HCC);
  k_attM<<<1, 32, 0, stream>>>(We2, att_e2, attM);
  LAUNCH(k_as_ad, NN * HH, hbuf, att_s2, att_d2, asb, adb);
  LAUNCH(k_fill, NN * HH, (float*)amaxK, 0.f, (long)NN * HH);
  LAUNCH(k_fill, NN * HH, denom, 0.f, (long)NN * HH);
  LAUNCH(k_fill, NN * HCC, obuf, 0.f, (long)NN * HCC);
  LAUNCH(k_alpha, EAA, ei, eattr, emean, asb, adb, attM, alpha, amaxK);
  LAUNCH(k_expsum, EAA, ei, alpha, amaxK, denom);
  k_scatter<<<(EAA + 7) / 8, 256, 0, stream>>>(ei, hbuf, alpha, denom, obuf);

  // ---- global mean pool + final linear ----
  LAUNCH(k_fill, GG * HCC, pooled, 0.f, (long)GG * HCC);
  LAUNCH(k_fill, GG, cnt, 0.f, (long)GG);
  LAUNCH(k_cnt, NN, batch, cnt);
  LAUNCH(k_pool, (long)NN * HCC, obuf, b2, batch, pooled);
  LAUNCH(k_head, GG * 2, pooled, cnt, lin_w, lin_b, outp);
}
